// Decoder_25847113188088
// MI455X (gfx1250) — compile-verified, matrix operand order checked by reading, not run on hardware
//
#include <hip/hip_runtime.h>
#include <hip/hip_bf16.h>

typedef __attribute__((ext_vector_type(16))) _Float16 v16h;
typedef __attribute__((ext_vector_type(8)))  float    v8f;

union Frag16 { v16h h; unsigned u[8]; };

// ---------------------------------------------------------------------------
// gfx1250 async global->LDS copy (16B per lane), ASYNCcnt-tracked.
// Inline asm (portable across toolchains; builtin signature is toolchain-
// specific).  LDS address VGPR = low 32 bits of the flat address (ISA 10.2:
// LDS aperture maps addr[31:0] to the LDS offset).
// ---------------------------------------------------------------------------
__device__ inline void async_copy_b128(void* lds, const void* g) {
  unsigned loff = (unsigned)(unsigned long long)lds;
  unsigned long long ga = (unsigned long long)g;
  asm volatile("global_load_async_to_lds_b128 %0, %1, off"
               :: "v"(loff), "v"(ga) : "memory");
}

__device__ inline void wait_async0() {
#if defined(__has_builtin)
#if __has_builtin(__builtin_amdgcn_s_wait_asynccnt)
  __builtin_amdgcn_s_wait_asynccnt(0);
  return;
#else
  asm volatile("s_wait_asynccnt 0x0" ::: "memory");
  return;
#endif
#else
  asm volatile("s_wait_asynccnt 0x0" ::: "memory");
#endif
}

// ---------------------------------------------------------------------------
// WMMA fragment loaders (per CDNA5 ISA 7.12.2 layouts, wave32)
// ---------------------------------------------------------------------------
__device__ inline void load_frag_rowmajor(const _Float16* base, int ld, Frag16& f) {
  const int l  = threadIdx.x & 31;
  const int m  = l & 15;
  const int kb = (l < 16) ? 0 : 8;
  const _Float16* p = base + (size_t)m * ld;
#pragma unroll
  for (int v = 0; v < 8; ++v) {
    int k = (v < 4) ? (kb + 2 * v) : (16 + kb + 2 * (v - 4));
    f.u[v] = *(const unsigned*)(p + k);
  }
}

__device__ inline void load_frag_colpairs(const _Float16* base, int ld, Frag16& f) {
  const int l  = threadIdx.x & 31;
  const int n  = l & 15;
  const int kb = (l < 16) ? 0 : 8;
#pragma unroll
  for (int v = 0; v < 8; ++v) {
    int k = (v < 4) ? (kb + 2 * v) : (16 + kb + 2 * (v - 4));
    unsigned lo = *(const unsigned short*)(base + (size_t)k * ld + n);
    unsigned hi = *(const unsigned short*)(base + (size_t)(k + 1) * ld + n);
    f.u[v] = lo | (hi << 16);
  }
}

// ---------------------------------------------------------------------------
// GEMM: C(MxN,f32) = A(MxK,f16) * B(KxN,f16) + bias, epilogues:
//   EPI 0: f32 out = acc + bias
//   EPI 1: f32 out = acc + bias + res
//   EPI 2: f16 out = gelu(acc + bias)
// Block tile 128x64, 8 waves (4x2) of 32x32 (2x2 WMMA frags), BK=32.
// Tiles staged with async global->LDS copies (ASYNCcnt), next tile prefetched.
// ---------------------------------------------------------------------------
template<int EPI>
__global__ __launch_bounds__(256) void gemm_wmma(
    const _Float16* __restrict__ A, const _Float16* __restrict__ B,
    const float* __restrict__ bias, const float* __restrict__ res,
    float* __restrict__ outf, _Float16* __restrict__ outh,
    int M, int N, int K)
{
  __shared__ __align__(16) _Float16 As[128 * 32];
  __shared__ __align__(16) _Float16 Bs[32 * 64];
  const int tid = threadIdx.x;
  const int wid = tid >> 5;
  const int wr  = wid & 3;
  const int wc  = wid >> 2;
  const int m0  = blockIdx.y * 128;
  const int n0  = blockIdx.x * 64;

  // per-thread staging coordinates
  const int i0 = tid, i1 = tid + 256;           // A: 512x16B (128 rows x 4)
  const int r0 = i0 >> 2, c0 = (i0 & 3) * 8;
  const int r1 = i1 >> 2, c1 = (i1 & 3) * 8;
  const int rb = tid >> 3, cb = (tid & 7) * 8;  // B: 256x16B (32 rows x 8)

  v8f acc[2][2] = {};

  for (int k0 = 0; k0 < K; k0 += 32) {
    async_copy_b128((char*)As + (size_t)i0 * 16,
                    A + (size_t)(m0 + r0) * K + k0 + c0);
    async_copy_b128((char*)As + (size_t)i1 * 16,
                    A + (size_t)(m0 + r1) * K + k0 + c1);
    async_copy_b128((char*)Bs + (size_t)tid * 16,
                    B + (size_t)(k0 + rb) * N + n0 + cb);
    if (k0 + 32 < K) {
      __builtin_prefetch(A + (size_t)(m0 + r0) * K + k0 + 32 + c0, 0, 1);
      __builtin_prefetch(B + (size_t)(k0 + 32 + rb) * N + n0 + cb, 0, 1);
    }
    wait_async0();
    __syncthreads();

    Frag16 fa[2], fb[2];
    load_frag_rowmajor(As + (wr * 32) * 32,        32, fa[0]);
    load_frag_rowmajor(As + (wr * 32 + 16) * 32,   32, fa[1]);
    load_frag_colpairs(Bs + wc * 32,               64, fb[0]);
    load_frag_colpairs(Bs + wc * 32 + 16,          64, fb[1]);
#pragma unroll
    for (int i = 0; i < 2; ++i)
#pragma unroll
      for (int j = 0; j < 2; ++j)
        acc[i][j] = __builtin_amdgcn_wmma_f32_16x16x32_f16(
            false, fa[i].h, false, fb[j].h, (short)0, acc[i][j], false, false);
    __syncthreads();
  }

  const int l   = tid & 31;
  const int nl  = l & 15;
  const int rb8 = (l < 16) ? 0 : 8;
#pragma unroll
  for (int i = 0; i < 2; ++i) {
#pragma unroll
    for (int j = 0; j < 2; ++j) {
      int col = n0 + wc * 32 + j * 16 + nl;
#pragma unroll
      for (int v = 0; v < 8; ++v) {
        int row = m0 + wr * 32 + i * 16 + rb8 + v;
        size_t oi = (size_t)row * N + col;
        float val = acc[i][j][v] + bias[col];
        if (EPI == 1) val += res[oi];
        if (EPI == 2) {
          val = 0.5f * val * (1.0f + erff(val * 0.70710678118f));
          outh[oi] = (_Float16)val;
        } else {
          outf[oi] = val;
        }
      }
    }
  }
}

// ---------------------------------------------------------------------------
// Fused attention: one wave per (q-block of 16 rows, head, batch).
// q,k,v: (4,16,1024,32) f16, already LN'd (q,k).  o: (4,1024,512) f16.
// ---------------------------------------------------------------------------
__global__ __launch_bounds__(32) void attn_kernel(
    const _Float16* __restrict__ q, const _Float16* __restrict__ k,
    const _Float16* __restrict__ v, _Float16* __restrict__ o)
{
  __shared__ float     sc[16 * 1024];   // 64 KB scores
  __shared__ _Float16  pk[16 * 1024];   // 32 KB packed probs
  __shared__ float     rinv[16];
  const int qb = blockIdx.x, h = blockIdx.y, b = blockIdx.z;
  const int l = threadIdx.x;
  const size_t hb = (size_t)(b * 16 + h) * 1024 * 32;
  const _Float16* Q  = q + hb;
  const _Float16* Kp = k + hb;
  const _Float16* Vp = v + hb;

  Frag16 fq;
  load_frag_rowmajor(Q + qb * 16 * 32, 32, fq);

  const int nl  = l & 15;
  const int rb8 = (l < 16) ? 0 : 8;

  // S = Q K^T / sqrt(32): B-frag of K^T == A-layout load of K rows.
  for (int j = 0; j < 64; ++j) {
    Frag16 fk;
    load_frag_rowmajor(Kp + j * 16 * 32, 32, fk);
    v8f s = {};
    s = __builtin_amdgcn_wmma_f32_16x16x32_f16(false, fq.h, false, fk.h,
                                               (short)0, s, false, false);
#pragma unroll
    for (int vv = 0; vv < 8; ++vv)
      sc[(rb8 + vv) * 1024 + j * 16 + nl] = s[vv] * 0.17677669529f;
  }
  __syncthreads();

  {   // softmax: 2 lanes per row, 512 cols each, combine via shfl_xor(16)
    const int row = l & 15;
    const int off = (l >> 4) * 512;
    float mx = -1e30f;
    for (int t = 0; t < 512; ++t) mx = fmaxf(mx, sc[row * 1024 + off + t]);
    mx = fmaxf(mx, __shfl_xor(mx, 16));
    float sum = 0.f;
    for (int t = 0; t < 512; ++t) {
      float e = __expf(sc[row * 1024 + off + t] - mx);
      sc[row * 1024 + off + t] = e;
      sum += e;
    }
    sum += __shfl_xor(sum, 16);
    if (l < 16) rinv[row] = 1.0f / sum;
  }
  __syncthreads();
  for (int r = 0; r < 16; ++r) {
    float inv = rinv[r];
    for (int t = 0; t < 32; ++t) {
      int c = t * 32 + l;
      pk[r * 1024 + c] = (_Float16)(sc[r * 1024 + c] * inv);
    }
  }
  __syncthreads();

  // O = P V : 32 K-chunks, two 16-col B frags covering d=0..31.
  v8f oacc[2] = {};
  for (int jj = 0; jj < 32; ++jj) {
    Frag16 fp, fv0, fv1;
    load_frag_rowmajor(pk + jj * 32, 1024, fp);
    load_frag_colpairs(Vp + (size_t)jj * 32 * 32,      32, fv0);
    load_frag_colpairs(Vp + (size_t)jj * 32 * 32 + 16, 32, fv1);
    oacc[0] = __builtin_amdgcn_wmma_f32_16x16x32_f16(false, fp.h, false, fv0.h,
                                                     (short)0, oacc[0], false, false);
    oacc[1] = __builtin_amdgcn_wmma_f32_16x16x32_f16(false, fp.h, false, fv1.h,
                                                     (short)0, oacc[1], false, false);
  }
#pragma unroll
  for (int t = 0; t < 2; ++t)
#pragma unroll
    for (int vv = 0; vv < 8; ++vv) {
      int row = qb * 16 + rb8 + vv;
      int col = h * 32 + t * 16 + nl;
      o[((size_t)b * 1024 + row) * 512 + col] = (_Float16)oacc[t][vv];
    }
}

// ---------------------------------------------------------------------------
// LayerNorm over 512 cols, wave per row, f16 output (GEMM operand).
// ---------------------------------------------------------------------------
template<bool AFF>
__global__ __launch_bounds__(128) void ln512_f16(
    const float* __restrict__ x, const float* __restrict__ g,
    const float* __restrict__ bta, _Float16* __restrict__ out, float eps)
{
  const int wid = threadIdx.x >> 5;
  const int l   = threadIdx.x & 31;
  const int row = blockIdx.x * 4 + wid;
  const float* xr = x + (size_t)row * 512;
  float vals[16];
  float s = 0.f;
#pragma unroll
  for (int t = 0; t < 16; ++t) { vals[t] = xr[t * 32 + l]; s += vals[t]; }
#pragma unroll
  for (int m = 16; m >= 1; m >>= 1) s += __shfl_xor(s, m);
  float mu = s * (1.0f / 512.0f);
  float var = 0.f;
#pragma unroll
  for (int t = 0; t < 16; ++t) { float d = vals[t] - mu; var += d * d; }
#pragma unroll
  for (int m = 16; m >= 1; m >>= 1) var += __shfl_xor(var, m);
  float rs = rsqrtf(var * (1.0f / 512.0f) + eps);
  _Float16* orow = out + (size_t)row * 512;
#pragma unroll
  for (int t = 0; t < 16; ++t) {
    int c = t * 32 + l;
    float y = (vals[t] - mu) * rs;
    if (AFF) y = y * g[c] + bta[c];
    orow[c] = (_Float16)y;
  }
}

// qkv (4,1024,1536) f32 -> q,k,v (4,16,1024,32) f16, LN(d=32) on q,k.
__global__ __launch_bounds__(128) void qkv_split_ln(
    const float* __restrict__ qkv,
    const float* __restrict__ nqg, const float* __restrict__ nqb,
    const float* __restrict__ nkg, const float* __restrict__ nkb,
    _Float16* __restrict__ q, _Float16* __restrict__ k, _Float16* __restrict__ v)
{
  const int wid = threadIdx.x >> 5;
  const int l   = threadIdx.x & 31;
  int id = blockIdx.x * 4 + wid;            // (b*1024+n)*16 + h
  int h  = id & 15;
  int bn = id >> 4;
  int b  = bn >> 10;
  int n  = bn & 1023;
  const float* base = qkv + (size_t)bn * 1536 + h * 32 + l;
  float qv = base[0];
  float kv = base[512];
  float vv = base[1024];
  auto wsum = [](float x) {
#pragma unroll
    for (int m = 16; m >= 1; m >>= 1) x += __shfl_xor(x, m);
    return x;
  };
  float qm = wsum(qv) * (1.f / 32.f);
  float qd = qv - qm;
  float qvar = wsum(qd * qd) * (1.f / 32.f);
  float qn = qd * rsqrtf(qvar + 1e-5f) * nqg[l] + nqb[l];
  float km = wsum(kv) * (1.f / 32.f);
  float kd = kv - km;
  float kvar = wsum(kd * kd) * (1.f / 32.f);
  float kn = kd * rsqrtf(kvar + 1e-5f) * nkg[l] + nkb[l];
  size_t oidx = (((size_t)(b * 16 + h)) * 1024 + n) * 32 + l;
  q[oidx] = (_Float16)qn;
  k[oidx] = (_Float16)kn;
  v[oidx] = (_Float16)vv;
}

// x + 2d sincos pos-embed (computed analytically). C=512, grid 32x32.
__global__ void add_pe(const float* __restrict__ x, float* __restrict__ y, int total)
{
  int idx = blockIdx.x * 256 + threadIdx.x;
  if (idx >= total) return;
  int cch  = idx & 511;
  int lpos = (idx >> 9) & 1023;
  int i = lpos >> 5, j = lpos & 31;
  int cc = cch & 255;
  float pos = (cch < 256) ? (float)j : (float)i;
  int m = cc & 127;
  float omega = __powf(10000.0f, -(float)m * (1.0f / 128.0f));
  float a = pos * omega;
  float pe = (cc < 128) ? __sinf(a) : __cosf(a);
  y[idx] = x[idx] + pe;
}

__global__ void cvt_f32_f16(const float* __restrict__ in, _Float16* __restrict__ out, int n)
{
  int i = blockIdx.x * 256 + threadIdx.x;
  if (i < n) out[i] = (_Float16)in[i];
}

// (4,1024,1024) -> (4,64,128,128) unpatchify (P=4, OUT_C=64)
__global__ void unpatch(const float* __restrict__ yb, float* __restrict__ out)
{
  int idx = blockIdx.x * 256 + threadIdx.x;
  if (idx >= 4 * 64 * 128 * 128) return;
  int x = idx & 127;
  int y = (idx >> 7) & 127;
  int c = (idx >> 14) & 63;
  int b = idx >> 20;
  int hh = y >> 2, p = y & 3, ww = x >> 2, qq = x & 3;
  out[idx] = yb[(((size_t)b * 1024) + hh * 32 + ww) * 1024 + (c * 16 + p * 4 + qq)];
}

// direct 3x3 SAME conv, NCHW / OIHW, 128x128
template<int CIN>
__global__ void conv3x3(const float* __restrict__ in, const float* __restrict__ w,
                        const float* __restrict__ bias, float* __restrict__ out,
                        int Cout, int relu)
{
  int idx = blockIdx.x * 256 + threadIdx.x;
  int total = 4 * Cout * 128 * 128;
  if (idx >= total) return;
  int x = idx & 127, y = (idx >> 7) & 127;
  int co = (idx >> 14) % Cout;
  int b  = idx / (Cout << 14);
  float s = bias[co];
  for (int ci = 0; ci < CIN; ++ci) {
    const float* ip = in + ((size_t)(b * CIN + ci) << 14);
    const float* wp = w + ((size_t)(co * CIN + ci) * 9);
#pragma unroll
    for (int ky = 0; ky < 3; ++ky) {
      int iy = y + ky - 1;
      if (iy < 0 || iy > 127) continue;
#pragma unroll
      for (int kx = 0; kx < 3; ++kx) {
        int ix = x + kx - 1;
        if (ix < 0 || ix > 127) continue;
        s += ip[iy * 128 + ix] * wp[ky * 3 + kx];
      }
    }
  }
  if (relu) s = fmaxf(s, 0.f);
  out[idx] = s;
}

// ---------------------------------------------------------------------------
extern "C" void kernel_launch(void* const* d_in, const int* in_sizes, int n_in,
                              void* d_out, int out_size, void* d_ws, size_t ws_size,
                              hipStream_t stream)
{
  const float* x_in   = (const float*)d_in[0];
  const float* qkv_w  = (const float*)d_in[1];
  const float* qkv_b  = (const float*)d_in[2];
  const float* proj_w = (const float*)d_in[3];
  const float* proj_b = (const float*)d_in[4];
  const float* nq_g   = (const float*)d_in[5];
  const float* nq_b   = (const float*)d_in[6];
  const float* nk_g   = (const float*)d_in[7];
  const float* nk_b   = (const float*)d_in[8];
  const float* n1_g   = (const float*)d_in[9];
  const float* n1_b   = (const float*)d_in[10];
  const float* n2_g   = (const float*)d_in[11];
  const float* n2_b   = (const float*)d_in[12];
  const float* fc1_w  = (const float*)d_in[13];
  const float* fc1_b  = (const float*)d_in[14];
  const float* fc2_w  = (const float*)d_in[15];
  const float* fc2_b  = (const float*)d_in[16];
  const float* fin_w  = (const float*)d_in[17];
  const float* fin_b  = (const float*)d_in[18];
  const float* c1_w   = (const float*)d_in[19];
  const float* c1_b   = (const float*)d_in[20];
  const float* c2_w   = (const float*)d_in[21];
  const float* c2_b   = (const float*)d_in[22];

  char* ws = (char*)d_ws;
  size_t off = 0;
  auto alloc = [&](size_t bytes) -> void* {
    void* p = ws + off;
    off = (off + bytes + 255) & ~(size_t)255;
    return p;
  };
  float*    x0      = (float*)   alloc(4ull * 1024 * 512 * 4);
  _Float16* qkvw16  = (_Float16*)alloc(8ull * 512 * 1536 * 2);
  _Float16* projw16 = (_Float16*)alloc(8ull * 512 * 512 * 2);
  _Float16* fc1w16  = (_Float16*)alloc(8ull * 512 * 2048 * 2);
  _Float16* fc2w16  = (_Float16*)alloc(8ull * 2048 * 512 * 2);
  _Float16* finw16  = (_Float16*)alloc(512ull * 1024 * 2);
  _Float16* h16     = (_Float16*)alloc(4096ull * 512 * 2);
  float*    qkvb    = (float*)   alloc(4096ull * 1536 * 4);
  _Float16* q16     = (_Float16*)alloc(4096ull * 512 * 2);
  _Float16* k16     = (_Float16*)alloc(4096ull * 512 * 2);
  _Float16* v16     = (_Float16*)alloc(4096ull * 512 * 2);
  _Float16* o16     = (_Float16*)alloc(4096ull * 512 * 2);
  _Float16* mid16   = (_Float16*)alloc(4096ull * 2048 * 2);
  float*    yb      = (float*)   alloc(4096ull * 1024 * 4);
  float*    up      = (float*)   alloc(4ull * 64 * 128 * 128 * 4);
  float*    c1o     = (float*)   alloc(4ull * 16 * 128 * 128 * 4);

  auto cvt = [&](const float* src, _Float16* dst, int n) {
    cvt_f32_f16<<<(n + 255) / 256, 256, 0, stream>>>(src, dst, n);
  };
  cvt(qkv_w, qkvw16, 8 * 512 * 1536);
  cvt(proj_w, projw16, 8 * 512 * 512);
  cvt(fc1_w, fc1w16, 8 * 512 * 2048);
  cvt(fc2_w, fc2w16, 8 * 2048 * 512);
  cvt(fin_w, finw16, 512 * 1024);

  int tot = 4 * 1024 * 512;
  add_pe<<<(tot + 255) / 256, 256, 0, stream>>>(x_in, x0, tot);

  for (int i = 0; i < 8; ++i) {
    ln512_f16<true><<<1024, 128, 0, stream>>>(x0, n1_g + i * 512, n1_b + i * 512, h16, 1e-5f);
    gemm_wmma<0><<<dim3(1536 / 64, 4096 / 128), 256, 0, stream>>>(
        h16, qkvw16 + (size_t)i * 512 * 1536, qkv_b + i * 1536, nullptr,
        qkvb, nullptr, 4096, 1536, 512);
    qkv_split_ln<<<16384, 128, 0, stream>>>(
        qkvb, nq_g + i * 32, nq_b + i * 32, nk_g + i * 32, nk_b + i * 32,
        q16, k16, v16);
    attn_kernel<<<dim3(64, 16, 4), 32, 0, stream>>>(q16, k16, v16, o16);
    gemm_wmma<1><<<dim3(512 / 64, 4096 / 128), 256, 0, stream>>>(
        o16, projw16 + (size_t)i * 512 * 512, proj_b + i * 512, x0,
        x0, nullptr, 4096, 512, 512);
    ln512_f16<true><<<1024, 128, 0, stream>>>(x0, n2_g + i * 512, n2_b + i * 512, h16, 1e-5f);
    gemm_wmma<2><<<dim3(2048 / 64, 4096 / 128), 256, 0, stream>>>(
        h16, fc1w16 + (size_t)i * 512 * 2048, fc1_b + i * 2048, nullptr,
        nullptr, mid16, 4096, 2048, 512);
    gemm_wmma<1><<<dim3(512 / 64, 4096 / 128), 256, 0, stream>>>(
        mid16, fc2w16 + (size_t)i * 2048 * 512, fc2_b + i * 512, x0,
        x0, nullptr, 4096, 512, 2048);
  }

  ln512_f16<false><<<1024, 128, 0, stream>>>(x0, nullptr, nullptr, h16, 1e-6f);
  gemm_wmma<0><<<dim3(1024 / 64, 4096 / 128), 256, 0, stream>>>(
      h16, finw16, fin_b, nullptr, yb, nullptr, 4096, 1024, 512);

  int utot = 4 * 64 * 128 * 128;
  unpatch<<<(utot + 255) / 256, 256, 0, stream>>>(yb, up);
  int c1tot = 4 * 16 * 128 * 128;
  conv3x3<64><<<(c1tot + 255) / 256, 256, 0, stream>>>(up, c1_w, c1_b, c1o, 16, 1);
  int c2tot = 4 * 3 * 128 * 128;
  conv3x3<16><<<(c2tot + 255) / 256, 256, 0, stream>>>(c1o, c2_w, c2_b, (float*)d_out, 3, 0);
}
